// SNNFinanceModel_8864812499629
// MI455X (gfx1250) — matrix operand order repaired and checked by reading, not compiled, not run on hardware
//
#include <hip/hip_runtime.h>
#include <hip/hip_bf16.h>

// ---------------------------------------------------------------------------
// Model constants (from reference)
// ---------------------------------------------------------------------------
#define BB     8          // batch
#define SS     256        // sequence
#define F_IN   987
#define DD     256        // model dim
#define DFF    768
#define LL     12         // layers
#define KBITS  16
#define TT     (SS*KBITS) // 4096 time steps
#define MM     (TT*BB)    // 32768 GEMM rows
#define BD     (BB*DD)    // 2048 parallel recurrence channels
#define SEG    16         // time steps per scan segment
#define NSEG   (TT/SEG)   // 256 segments
#define QMAXF  65535.0f
#define BLOCK_VTH 0.05f
#define FFN_VTH   0.15f

typedef __attribute__((ext_vector_type(16))) _Float16 v16h;
typedef __attribute__((ext_vector_type(8)))  _Float16 v8h;
typedef __attribute__((ext_vector_type(8)))  float    v8f;

__device__ __forceinline__ float sigmoidf_(float x){ return 1.0f/(1.0f+__expf(-x)); }
__device__ __forceinline__ float softplusf_(float x){ return (x > 20.f) ? x : log1pf(__expf(x)); }

// rotation coefficients for the resonate-and-fire recurrence (per layer, per d)
__device__ __forceinline__ void rotor_coeffs(const float* bb, const float* ba, const float* bo,
                                             int layer, int d, float& c, float& sn){
  float decay = sigmoidf_(bb[layer*DD+d]) * __expf(-softplusf_(ba[layer*DD+d]));
  float om    = softplusf_(bo[layer*DD+d]);
  c = decay * __cosf(om);
  sn = decay * __sinf(om);
}

// ---------------------------------------------------------------------------
// WMMA fragment helpers (CDNA5 wave32, V_WMMA_F32_16X16X32_F16 layouts)
// A 16x32 f16: lane l holds row M=l%16; halves 0..15 -> K=kk+(l/16)*16..+15
//              (two 16B contiguous loads)
// B 32x16 f16 from TRANSPOSED weights Bt[N][K]: lane l holds col N=l%16,
//              K=kk+(l/16)*16..+15 (one 32B contiguous load)
// C/D 16x16 f32: lane l, vgpr j -> row = 8*(l/16)+j, col = l%16
// ---------------------------------------------------------------------------
__device__ __forceinline__ v16h load_a16x32(const _Float16* __restrict__ A, int lda,
                                            int rowBase, int kk, int lane){
  const _Float16* p = A + (size_t)(rowBase + (lane & 15)) * lda + kk + ((lane >> 4) << 4);
  v8h lo = *(const v8h*)p;
  v8h hi = *(const v8h*)(p + 8);
  v16h a;
  #pragma unroll
  for (int i = 0; i < 8; ++i){ a[i] = lo[i]; a[i+8] = hi[i]; }
  return a;
}

__device__ __forceinline__ v16h load_b32x16t(const _Float16* __restrict__ Bt, int Kd,
                                             int colBase, int kk, int lane){
  const _Float16* p = Bt + (size_t)(colBase + (lane & 15)) * Kd + kk + ((lane >> 4) << 4);
  return *(const v16h*)p;
}

#define WMMA16x16x32(a,b,c) __builtin_amdgcn_wmma_f32_16x16x32_f16(false,(a),false,(b),(short)0,(c),false,false)

// ---------------------------------------------------------------------------
// Weight conversion: f32 (L,K,N) -> f16 transposed (L,N,K)
// ---------------------------------------------------------------------------
__global__ void conv_transpose_w(const float* __restrict__ src, _Float16* __restrict__ dst,
                                 int Kd, int Nd, int total){
  for (int idx = blockIdx.x * blockDim.x + threadIdx.x; idx < total;
       idx += gridDim.x * blockDim.x){
    int kn  = Kd * Nd;
    int l   = idx / kn;
    int rem = idx - l * kn;
    int k   = rem / Nd;
    int n   = rem - k * Nd;
    dst[(size_t)l * kn + (size_t)n * Kd + k] = (_Float16)src[idx];
  }
}

// ---------------------------------------------------------------------------
// Input projection: proj[r,d] = sum_f features[r,f] * ipw[d,f]   (r = b*S+s)
// ---------------------------------------------------------------------------
__global__ void proj_kernel(const float* __restrict__ features, const float* __restrict__ ipw,
                            float* __restrict__ proj){
  int idx = blockIdx.x * blockDim.x + threadIdx.x;   // BB*SS*DD threads
  int r = idx >> 8, d = idx & 255;
  const float* fr = features + (size_t)r * F_IN;
  const float* wr = ipw + (size_t)d * F_IN;
  float acc = 0.f;
  for (int f = 0; f < F_IN; ++f) acc = fmaf(fr[f], wr[f], acc);
  proj[idx] = acc;
}

// ---------------------------------------------------------------------------
// fp16_encode: quantize to 16 MSB-first bit planes -> X (T,B,D) f16 {0,1}
// ---------------------------------------------------------------------------
__global__ void encode_kernel(const float* __restrict__ proj, _Float16* __restrict__ X){
  int idx = blockIdx.x * blockDim.x + threadIdx.x;   // BB*SS*DD threads
  int r = idx >> 8, d = idx & 255;
  int b = r >> 8, s = r & 255;                       // S == 256
  float x  = proj[idx];
  float rr = (tanhf(x) + 1.0f) * 0.5f;
  int q = (int)roundf(rr * QMAXF);
  q = q < 0 ? 0 : (q > 65535 ? 65535 : q);
  #pragma unroll
  for (int k = 0; k < KBITS; ++k){
    int bit = (q >> (15 - k)) & 1;
    int t   = (s << 4) + k;
    X[((size_t)(t * BB + b) << 8) + d] = (_Float16)bit;
  }
}

// ---------------------------------------------------------------------------
// Dual GEMM, 16x32 per wave per matrix (4 accumulators, A reused 4x):
// U = X @ W_in ; G = sigmoid(X @ W_gate)       (M x 256) x (256 x 256)
// ---------------------------------------------------------------------------
__global__ void layer_in_gemm(const _Float16* __restrict__ X,
                              const _Float16* __restrict__ WtIn,
                              const _Float16* __restrict__ WtGate,
                              _Float16* __restrict__ U, _Float16* __restrict__ G){
  const int NT2 = DD / 32;                             // 8 column-pairs
  int lane = threadIdx.x & 31;
  int tid  = blockIdx.x * 8 + (threadIdx.x >> 5);
  int bm = tid / NT2, nb = (tid - bm * NT2) * 32;
  v8f accU[2] = {{},{}};
  v8f accG[2] = {{},{}};
  #pragma unroll
  for (int kk = 0; kk < DD; kk += 32){
    if (kk + 32 < DD){
      __builtin_prefetch(WtIn   + (size_t)nb * DD + kk + 32, 0, 1);
      __builtin_prefetch(WtGate + (size_t)nb * DD + kk + 32, 0, 1);
    }
    v16h a = load_a16x32(X, DD, bm * 16, kk, lane);
    #pragma unroll
    for (int t = 0; t < 2; ++t){
      v16h b0 = load_b32x16t(WtIn,   DD, nb + t * 16, kk, lane);
      accU[t] = WMMA16x16x32(a, b0, accU[t]);
      v16h b1 = load_b32x16t(WtGate, DD, nb + t * 16, kk, lane);
      accG[t] = WMMA16x16x32(a, b1, accG[t]);
    }
  }
  int row0 = bm * 16 + ((lane >> 4) << 3);
  #pragma unroll
  for (int t = 0; t < 2; ++t){
    int col = nb + t * 16 + (lane & 15);
    #pragma unroll
    for (int j = 0; j < 8; ++j){
      size_t o = (size_t)(row0 + j) * DD + col;
      U[o] = (_Float16)accU[t][j];
      G[o] = (_Float16)sigmoidf_(accG[t][j]);
    }
  }
}

// ---------------------------------------------------------------------------
// Parallel scan for the linear resonate-and-fire recurrence.
// Stage A: per-segment local result from zero state (fully coalesced).
// ---------------------------------------------------------------------------
__global__ void scan_local_kernel(const _Float16* __restrict__ U, float2* __restrict__ VSEG,
                                  const float* __restrict__ bb, const float* __restrict__ ba,
                                  const float* __restrict__ bo, int layer){
  int idx = blockIdx.x * blockDim.x + threadIdx.x;   // NSEG*BD threads
  int bd = idx & (BD - 1), seg = idx >> 11;
  float c, sn; rotor_coeffs(bb, ba, bo, layer, bd & 255, c, sn);
  float re = 0.f, im = 0.f;
  int t0 = seg * SEG;
  #pragma unroll
  for (int j = 0; j < SEG; ++j){
    float u   = (float)U[(size_t)(t0 + j) * BD + bd];
    float re2 = fmaf(c, re, fmaf(-sn, im, u));
    float im2 = fmaf(sn, re, c * im);
    re = re2; im = im2;
  }
  VSEG[idx] = make_float2(re, im);                   // idx = seg*BD + bd
}

// ---------------------------------------------------------------------------
// Stage B: Hillis-Steele scan over 256 segments per channel in LDS.
// Segment operator: s -> Cm*s + v, Cm = R^SEG (complex, constant) -> the
// running multiplier just squares each step. Writes EXCLUSIVE prefix state.
// ---------------------------------------------------------------------------
__global__ void scan_prefix_kernel(const float2* __restrict__ VSEG, float2* __restrict__ SSTART,
                                   const float* __restrict__ bb, const float* __restrict__ ba,
                                   const float* __restrict__ bo, int layer){
  __shared__ float2 sv[NSEG];
  int bd = blockIdx.x;                               // 0..BD-1
  int seg = threadIdx.x;                             // 0..NSEG-1
  float c, sn; rotor_coeffs(bb, ba, bo, layer, bd & 255, c, sn);
  // Cm = (c + i*sn)^SEG via 4 squarings (SEG = 16)
  float Mx = c, My = sn;
  #pragma unroll
  for (int q = 0; q < 4; ++q){
    float nx = Mx * Mx - My * My;
    My = 2.f * Mx * My;
    Mx = nx;
  }
  float2 x = VSEG[(size_t)seg * BD + bd];
  sv[seg] = x;
  __syncthreads();
  for (int ofs = 1; ofs < NSEG; ofs <<= 1){
    float2 prev = (seg >= ofs) ? sv[seg - ofs] : make_float2(0.f, 0.f);
    __syncthreads();
    x.x = fmaf(Mx, prev.x, fmaf(-My, prev.y, x.x));
    x.y = fmaf(Mx, prev.y, fmaf( My, prev.x, x.y));
    sv[seg] = x;
    float nx = Mx * Mx - My * My;                    // M <- M^2
    My = 2.f * Mx * My;
    Mx = nx;
    __syncthreads();
  }
  float2 e = (seg == 0) ? make_float2(0.f, 0.f) : sv[seg - 1];   // exclusive
  SSTART[(size_t)seg * BD + bd] = e;
}

// ---------------------------------------------------------------------------
// Stage C: re-run each segment from its prefix state; fuse spike * gate.
// ---------------------------------------------------------------------------
__global__ void scan_apply_kernel(const _Float16* __restrict__ U, _Float16* __restrict__ G,
                                  const float2* __restrict__ SSTART,
                                  const float* __restrict__ bb, const float* __restrict__ ba,
                                  const float* __restrict__ bo, int layer){
  int idx = blockIdx.x * blockDim.x + threadIdx.x;   // NSEG*BD threads
  int bd = idx & (BD - 1), seg = idx >> 11;
  float c, sn; rotor_coeffs(bb, ba, bo, layer, bd & 255, c, sn);
  float2 s0 = SSTART[idx];
  float re = s0.x, im = s0.y;
  int t0 = seg * SEG;
  #pragma unroll
  for (int j = 0; j < SEG; ++j){
    size_t o  = (size_t)(t0 + j) * BD + bd;
    float u   = (float)U[o];
    float re2 = fmaf(c, re, fmaf(-sn, im, u));
    float im2 = fmaf(sn, re, c * im);
    re = re2; im = im2;
    float sp = (re2 >= BLOCK_VTH) ? 1.f : 0.f;
    G[o] = (_Float16)(sp * (float)G[o]);
  }
}

// ---------------------------------------------------------------------------
// H = SG @ W_out + Xres, 16x64 per wave (4 accumulators, A reused 4x)
// ---------------------------------------------------------------------------
__global__ void gemm_out_res(const _Float16* __restrict__ SG, const _Float16* __restrict__ Wt,
                             const _Float16* __restrict__ Xres, _Float16* __restrict__ H){
  const int NT4 = DD / 64;                             // 4
  int lane = threadIdx.x & 31;
  int tid  = blockIdx.x * 8 + (threadIdx.x >> 5);
  int bm = tid / NT4, nb = (tid - bm * NT4) * 64;
  v8f acc[4] = {{},{},{},{}};
  #pragma unroll
  for (int kk = 0; kk < DD; kk += 32){
    if (kk + 32 < DD) __builtin_prefetch(Wt + (size_t)nb * DD + kk + 32, 0, 1);
    v16h a = load_a16x32(SG, DD, bm * 16, kk, lane);
    #pragma unroll
    for (int t = 0; t < 4; ++t){
      v16h b = load_b32x16t(Wt, DD, nb + t * 16, kk, lane);
      acc[t] = WMMA16x16x32(a, b, acc[t]);
    }
  }
  int row0 = bm * 16 + ((lane >> 4) << 3);
  #pragma unroll
  for (int t = 0; t < 4; ++t){
    int col = nb + t * 16 + (lane & 15);
    #pragma unroll
    for (int j = 0; j < 8; ++j){
      size_t o = (size_t)(row0 + j) * DD + col;
      H[o] = (_Float16)(acc[t][j] + (float)Xres[o]);
    }
  }
}

// ---------------------------------------------------------------------------
// FFN gate/up fused, 16x32 per wave per matrix (4 accumulators):
// F = spike(silu(H@Wg) * (H@Wu) - vth)          (M x 256) x (256 x 768)
// ---------------------------------------------------------------------------
__global__ void ffn_gate_up(const _Float16* __restrict__ H,
                            const _Float16* __restrict__ WtG, const _Float16* __restrict__ WtU,
                            _Float16* __restrict__ F){
  const int NT2 = DFF / 32;                            // 24
  int lane = threadIdx.x & 31;
  int tid  = blockIdx.x * 8 + (threadIdx.x >> 5);
  int bm = tid / NT2, nb = (tid - bm * NT2) * 32;
  v8f accG[2] = {{},{}};
  v8f accU[2] = {{},{}};
  #pragma unroll
  for (int kk = 0; kk < DD; kk += 32){
    if (kk + 32 < DD){
      __builtin_prefetch(WtG + (size_t)nb * DD + kk + 32, 0, 1);
      __builtin_prefetch(WtU + (size_t)nb * DD + kk + 32, 0, 1);
    }
    v16h a = load_a16x32(H, DD, bm * 16, kk, lane);
    #pragma unroll
    for (int t = 0; t < 2; ++t){
      v16h bg = load_b32x16t(WtG, DD, nb + t * 16, kk, lane);
      accG[t] = WMMA16x16x32(a, bg, accG[t]);
      v16h bu = load_b32x16t(WtU, DD, nb + t * 16, kk, lane);
      accU[t] = WMMA16x16x32(a, bu, accU[t]);
    }
  }
  int row0 = bm * 16 + ((lane >> 4) << 3);
  #pragma unroll
  for (int t = 0; t < 2; ++t){
    int col = nb + t * 16 + (lane & 15);
    #pragma unroll
    for (int j = 0; j < 8; ++j){
      float g  = accG[t][j];
      float uu = accU[t][j];
      float v  = (g * sigmoidf_(g)) * uu;              // silu(g)*u
      F[(size_t)(row0 + j) * DFF + col] = (_Float16)((v >= FFN_VTH) ? 1.f : 0.f);
    }
  }
}

// ---------------------------------------------------------------------------
// Xnext = H + F @ Wd, 16x64 per wave          (M x 768) x (768 x 256)
// ---------------------------------------------------------------------------
__global__ void ffn_down_res(const _Float16* __restrict__ F, const _Float16* __restrict__ WtD,
                             const _Float16* __restrict__ H, _Float16* __restrict__ Xn){
  const int NT4 = DD / 64;                             // 4
  int lane = threadIdx.x & 31;
  int tid  = blockIdx.x * 8 + (threadIdx.x >> 5);
  int bm = tid / NT4, nb = (tid - bm * NT4) * 64;
  v8f acc[4] = {{},{},{},{}};
  #pragma unroll
  for (int kk = 0; kk < DFF; kk += 32){
    if (kk + 32 < DFF) __builtin_prefetch(WtD + (size_t)nb * DFF + kk + 32, 0, 1);
    v16h a = load_a16x32(F, DFF, bm * 16, kk, lane);
    #pragma unroll
    for (int t = 0; t < 4; ++t){
      v16h b = load_b32x16t(WtD, DFF, nb + t * 16, kk, lane);
      acc[t] = WMMA16x16x32(a, b, acc[t]);
    }
  }
  int row0 = bm * 16 + ((lane >> 4) << 3);
  #pragma unroll
  for (int t = 0; t < 4; ++t){
    int col = nb + t * 16 + (lane & 15);
    #pragma unroll
    for (int j = 0; j < 8; ++j){
      size_t o = (size_t)(row0 + j) * DD + col;
      Xn[o] = (_Float16)(acc[t][j] + (float)H[o]);
    }
  }
}

// ---------------------------------------------------------------------------
// RMSNorm over D per row
// ---------------------------------------------------------------------------
__global__ void rmsnorm_kernel(const _Float16* __restrict__ Xf, const float* __restrict__ w,
                               _Float16* __restrict__ O){
  __shared__ float sd[DD];
  int m = blockIdx.x, e = threadIdx.x;
  float x = (float)Xf[(size_t)m * DD + e];
  sd[e] = x * x;
  __syncthreads();
  for (int s = DD / 2; s > 0; s >>= 1){
    if (e < s) sd[e] += sd[e + s];
    __syncthreads();
  }
  float inv = rsqrtf(sd[0] / (float)DD + 1e-6f);
  O[(size_t)m * DD + e] = (_Float16)(x * inv * w[e]);
}

// ---------------------------------------------------------------------------
// PLIF scan + fp16_decode fused (nonlinear soft reset -> stays sequential):
// dec[b,s,d] = sum_k spike_k * 2^(15-k)/QMAX
// ---------------------------------------------------------------------------
__global__ void plif_decode_kernel(const _Float16* __restrict__ Hn,
                                   const float* __restrict__ plif_w,
                                   const float* __restrict__ plif_vth,
                                   float* __restrict__ dec){
  int tid = blockIdx.x * blockDim.x + threadIdx.x;   // BD threads: tid = b*256+d
  int b = tid >> 8, d = tid & 255;
  float beta = sigmoidf_(plif_w[d]);
  float vth  = plif_vth[d];
  float v = 0.f;
  for (int s = 0; s < SS; ++s){
    float acc = 0.f;
    #pragma unroll
    for (int k = 0; k < KBITS; ++k){
      int t = (s << 4) + k;
      float h    = (float)Hn[(size_t)t * BD + tid];
      float vpre = fmaf(beta, v, (1.f - beta) * h);
      float spk  = ((vpre - vth) >= 0.f) ? 1.f : 0.f;
      v = vpre - spk * vth;
      acc = fmaf(spk, (float)(1 << (15 - k)) / QMAXF, acc);
    }
    dec[((size_t)b * SS + s) * DD + d] = acc;
  }
}

// ---------------------------------------------------------------------------
// h2 = norm_gain * (dec @ decode_proj_w^T + b - mean_e)
// ---------------------------------------------------------------------------
__global__ void h2_kernel(const float* __restrict__ dec, const float* __restrict__ Wdec,
                          const float* __restrict__ bdec, const float* __restrict__ gain,
                          float* __restrict__ h2){
  __shared__ float sd[DD];
  int r = blockIdx.x, e = threadIdx.x;
  const float* dr = dec  + (size_t)r * DD;
  const float* wr = Wdec + (size_t)e * DD;
  float acc = bdec[e];
  for (int d = 0; d < DD; ++d) acc = fmaf(dr[d], wr[d], acc);
  sd[e] = acc;
  __syncthreads();
  for (int s = DD / 2; s > 0; s >>= 1){
    if (e < s) sd[e] += sd[e + s];
    __syncthreads();
  }
  float mean = sd[0] / (float)DD;
  h2[(size_t)r * DD + e] = gain[e] * (acc - mean);
}

// ---------------------------------------------------------------------------
// Head: 4 outputs per (b,s) + nonlinear transforms -> decisions
// ---------------------------------------------------------------------------
__global__ void head_kernel(const float* __restrict__ h2, const float* __restrict__ hw,
                            const float* __restrict__ hb, float* __restrict__ out){
  int r = blockIdx.x * blockDim.x + threadIdx.x;     // 2048
  const float* row = h2 + (size_t)r * DD;
  float raw[4];
  #pragma unroll
  for (int o = 0; o < 4; ++o){
    float acc = hb[o];
    const float* w = hw + o * DD;
    for (int d = 0; d < DD; ++d) acc = fmaf(row[d], w[d], acc);
    raw[o] = acc;
  }
  out[r * 4 + 0] = tanhf(raw[0]);
  out[r * 4 + 1] = 1.0f + 9.0f * sigmoidf_(raw[1]);            // 1 + (MAX_LEV-1)*sig
  out[r * 4 + 2] = 0.005f + 0.095f * sigmoidf_(raw[2]);        // SL_MIN..SL_MAX
  out[r * 4 + 3] = 0.01f  + 0.19f  * sigmoidf_(raw[3]);        // TP_MIN..TP_MAX
}

// ---------------------------------------------------------------------------
// Reconstruction: rec[r,f] = sum_d h2[r,d] * ipw[d,f]
// ---------------------------------------------------------------------------
__global__ void recon_kernel(const float* __restrict__ h2, const float* __restrict__ ipw,
                             float* __restrict__ rec, int total){
  int idx = blockIdx.x * blockDim.x + threadIdx.x;
  if (idx >= total) return;
  int r = idx / F_IN, f = idx - r * F_IN;
  const float* row = h2 + (size_t)r * DD;
  float acc = 0.f;
  for (int d = 0; d < DD; ++d) acc = fmaf(row[d], ipw[(size_t)d * F_IN + f], acc);
  rec[idx] = acc;
}

// ---------------------------------------------------------------------------
// Host orchestration
// ---------------------------------------------------------------------------
extern "C" void kernel_launch(void* const* d_in, const int* in_sizes, int n_in,
                              void* d_out, int out_size, void* d_ws, size_t ws_size,
                              hipStream_t stream){
  const float* features  = (const float*)d_in[0];
  const float* ipw       = (const float*)d_in[1];
  const float* norm_gain = (const float*)d_in[2];
  const float* dec_w     = (const float*)d_in[3];
  const float* dec_b     = (const float*)d_in[4];
  const float* head_w    = (const float*)d_in[5];
  const float* head_b    = (const float*)d_in[6];
  const float* out_nw    = (const float*)d_in[7];
  const float* plif_w    = (const float*)d_in[8];
  const float* plif_vth  = (const float*)d_in[9];
  const float* W_in      = (const float*)d_in[10];
  const float* W_gate    = (const float*)d_in[11];
  const float* W_out     = (const float*)d_in[12];
  const float* b_beta    = (const float*)d_in[13];
  const float* b_alpha   = (const float*)d_in[14];
  const float* b_omega   = (const float*)d_in[15];
  const float* Wg        = (const float*)d_in[16];
  const float* Wu        = (const float*)d_in[17];
  const float* Wd        = (const float*)d_in[18];

  // ---- workspace layout (all 256B aligned by construction) ----
  char* ws = (char*)d_ws;
  size_t off = 0;
  auto alloc = [&](size_t bytes)->char*{ char* p = ws + off; off += bytes; return p; };
  _Float16* WT_IN   = (_Float16*)alloc((size_t)LL * DD  * DD  * 2);
  _Float16* WT_GATE = (_Float16*)alloc((size_t)LL * DD  * DD  * 2);
  _Float16* WT_OUT  = (_Float16*)alloc((size_t)LL * DD  * DD  * 2);
  _Float16* WT_G    = (_Float16*)alloc((size_t)LL * DFF * DD  * 2);  // (l, n=DFF, k=DD)
  _Float16* WT_U    = (_Float16*)alloc((size_t)LL * DFF * DD  * 2);
  _Float16* WT_D    = (_Float16*)alloc((size_t)LL * DD  * DFF * 2);  // (l, n=DD, k=DFF)
  _Float16* BUF_X   = (_Float16*)alloc((size_t)MM * DD  * 2);
  _Float16* BUF_U   = (_Float16*)alloc((size_t)MM * DD  * 2);
  _Float16* BUF_G   = (_Float16*)alloc((size_t)MM * DD  * 2);
  _Float16* BUF_H   = (_Float16*)alloc((size_t)MM * DD  * 2);
  _Float16* BUF_F   = (_Float16*)alloc((size_t)MM * DFF * 2);
  float2*   VSEG    = (float2*)  alloc((size_t)NSEG * BD * sizeof(float2));
  float2*   SSTART  = (float2*)  alloc((size_t)NSEG * BD * sizeof(float2));
  float*    DECBUF  = (float*)   alloc((size_t)BB * SS * DD * 4);    // also proj scratch
  float*    H2BUF   = (float*)   alloc((size_t)BB * SS * DD * 4);
  (void)ws_size; (void)n_in; (void)in_sizes; (void)out_size;

  float* out_dec = (float*)d_out;              // (B,S,1,4) = 8192
  float* out_rec = out_dec + BB * SS * 4;      // (B,S,F_IN)

  // ---- weight conversion (f32 -> transposed f16) ----
  {
    int tot_dd  = LL * DD * DD;
    int tot_dff = LL * DD * DFF;
    conv_transpose_w<<<(tot_dd  + 255) / 256, 256, 0, stream>>>(W_in,   WT_IN,   DD,  DD,  tot_dd);
    conv_transpose_w<<<(tot_dd  + 255) / 256, 256, 0, stream>>>(W_gate, WT_GATE, DD,  DD,  tot_dd);
    conv_transpose_w<<<(tot_dd  + 255) / 256, 256, 0, stream>>>(W_out,  WT_OUT,  DD,  DD,  tot_dd);
    conv_transpose_w<<<(tot_dff + 255) / 256, 256, 0, stream>>>(Wg,     WT_G,    DD,  DFF, tot_dff);
    conv_transpose_w<<<(tot_dff + 255) / 256, 256, 0, stream>>>(Wu,     WT_U,    DD,  DFF, tot_dff);
    conv_transpose_w<<<(tot_dff + 255) / 256, 256, 0, stream>>>(Wd,     WT_D,    DFF, DD,  tot_dff);
  }

  // ---- input projection + spike encode ----
  proj_kernel  <<<(BB * SS * DD) / 256, 256, 0, stream>>>(features, ipw, DECBUF);
  encode_kernel<<<(BB * SS * DD) / 256, 256, 0, stream>>>(DECBUF, BUF_X);

  // ---- 12 layers ----
  const unsigned gemm_in_blocks   = (MM / 16) * (DD  / 32) / 8;   // 2048 (dual 16x32)
  const unsigned gemm_out_blocks  = (MM / 16) * (DD  / 64) / 8;   // 1024 (16x64)
  const unsigned gemm_ffn1_blocks = (MM / 16) * (DFF / 32) / 8;   // 6144 (dual 16x32)
  const unsigned scan_blocks      = (NSEG * BD) / 256;            // 2048
  for (int l = 0; l < LL; ++l){
    const _Float16* wtin = WT_IN   + (size_t)l * DD  * DD;
    const _Float16* wtga = WT_GATE + (size_t)l * DD  * DD;
    const _Float16* wtou = WT_OUT  + (size_t)l * DD  * DD;
    const _Float16* wtg  = WT_G    + (size_t)l * DFF * DD;
    const _Float16* wtu  = WT_U    + (size_t)l * DFF * DD;
    const _Float16* wtd  = WT_D    + (size_t)l * DD  * DFF;

    layer_in_gemm<<<gemm_in_blocks, 256, 0, stream>>>(BUF_X, wtin, wtga, BUF_U, BUF_G);
    scan_local_kernel <<<scan_blocks, 256, 0, stream>>>(BUF_U, VSEG, b_beta, b_alpha, b_omega, l);
    scan_prefix_kernel<<<BD, NSEG, 0, stream>>>(VSEG, SSTART, b_beta, b_alpha, b_omega, l);
    scan_apply_kernel <<<scan_blocks, 256, 0, stream>>>(BUF_U, BUF_G, SSTART, b_beta, b_alpha, b_omega, l);
    gemm_out_res <<<gemm_out_blocks, 256, 0, stream>>>(BUF_G, wtou, BUF_X, BUF_H);
    ffn_gate_up  <<<gemm_ffn1_blocks, 256, 0, stream>>>(BUF_H, wtg, wtu, BUF_F);
    ffn_down_res <<<gemm_out_blocks, 256, 0, stream>>>(BUF_F, wtd, BUF_H, BUF_X);
  }

  // ---- output stage ----
  rmsnorm_kernel    <<<MM, DD, 0, stream>>>(BUF_X, out_nw, BUF_U);
  plif_decode_kernel<<<BD / 256, 256, 0, stream>>>(BUF_U, plif_w, plif_vth, DECBUF);
  h2_kernel         <<<BB * SS, DD, 0, stream>>>(DECBUF, dec_w, dec_b, norm_gain, H2BUF);
  head_kernel       <<<(BB * SS) / 256, 256, 0, stream>>>(H2BUF, head_w, head_b, out_dec);
  {
    int total = BB * SS * F_IN;
    recon_kernel<<<(total + 255) / 256, 256, 0, stream>>>(H2BUF, ipw, out_rec, total);
  }
}
